// SegmentationModule_46334107189549
// MI455X (gfx1250) — compile-verified, matrix operand order checked by reading, not run on hardware
//
#include <hip/hip_runtime.h>
#include <hip/hip_bf16.h>

// ---------------------------------------------------------------------------
// PointNet++ SA + segmentation head for MI455X (gfx1250, wave32, WMMA).
//
// Roofline: ~22 GFLOP of GEMM vs ~0.5 GB of HBM traffic -> bandwidth bound
// (~25us at 23.3 TB/s) except FPS which is 1023 sequential steps (latency
// bound, one workgroup per batch). GEMMs use V_WMMA_F32_16X16X32_BF16 with
// activations stored N-major bf16 so each fragment half is one b128 load.
// GEMM is templated on (K, tile shape) so the K-loop fully unrolls:
// straight-line WMMA stream, double-buffered LDS with ONE barrier per K-step
// and next-tile global loads overlapping current-tile WMMAs.
// BN statistics use a deterministic two-pass reduction (no float atomics)
// so graph replays are bit-identical.
// ---------------------------------------------------------------------------

typedef __attribute__((ext_vector_type(16))) __bf16        v16bf;
typedef __attribute__((ext_vector_type(8)))  float         v8f;
typedef __attribute__((ext_vector_type(4)))  unsigned int  u32x4;
typedef __attribute__((ext_vector_type(4)))  float         f32x4;

#define NPTS    16384
#define NPOINT  1024
#define NSAMP   16
#define NBATCH  8
#define N0      (NBATCH * NPOINT * NSAMP)   // 131072 GEMM columns, layers 0-2
#define NPQ     (NBATCH * NPOINT)           // 8192  GEMM columns, conv layers
#define K0PAD   288                          // 259 padded to multiple of 32
#define RAD2    0.09f

static __device__ __forceinline__ unsigned short f2bf(float f) {
  unsigned u = __float_as_uint(f);
  unsigned r = u + 0x7fffu + ((u >> 16) & 1u);   // round-to-nearest-even
  return (unsigned short)(r >> 16);
}
static __device__ __forceinline__ float bf2f(unsigned short h) {
  return __uint_as_float(((unsigned)h) << 16);
}
static __device__ __forceinline__ v16bf make_v16bf(u32x4 a, u32x4 b) {
  union { u32x4 u[2]; v16bf v; } t;
  t.u[0] = a; t.u[1] = b;
  return t.v;
}

// ---------------------------------------------------------------------------
// FPS: one workgroup per batch, 1024 threads, 16 points/thread in registers.
// ---------------------------------------------------------------------------
__global__ __launch_bounds__(1024) void fps_kernel(const float* __restrict__ xyz,
                                                   int* __restrict__ fpsInds) {
  const int b = blockIdx.x;
  const int tid = threadIdx.x;
  const float* P = xyz + (size_t)b * NPTS * 3;

  float px[16], py[16], pz[16], d[16];
#pragma unroll
  for (int j = 0; j < 16; ++j) {
    int pi = tid * 16 + j;
    px[j] = P[pi * 3 + 0];
    py[j] = P[pi * 3 + 1];
    pz[j] = P[pi * 3 + 2];
    d[j] = 1e10f;
  }

  __shared__ float s_c[3];
  __shared__ float s_val[32];
  __shared__ int   s_idx[32];
  __shared__ int   s_last;

  if (tid == 0) {
    fpsInds[b * NPOINT] = 0;
    s_c[0] = P[0]; s_c[1] = P[1]; s_c[2] = P[2];
  }
  __syncthreads();

  for (int step = 1; step < NPOINT; ++step) {
    float cx = s_c[0], cy = s_c[1], cz = s_c[2];
    float bv = -1.f; int bi = 0x7fffffff;
#pragma unroll
    for (int j = 0; j < 16; ++j) {
      float dx = px[j] - cx, dy = py[j] - cy, dz = pz[j] - cz;
      float dd = dx * dx + dy * dy + dz * dz;
      d[j] = fminf(d[j], dd);
      int pi = tid * 16 + j;
      if (d[j] > bv || (d[j] == bv && pi < bi)) { bv = d[j]; bi = pi; }
    }
    // wave32 argmax (first-index tie break, matches jnp.argmax)
#pragma unroll
    for (int off = 16; off; off >>= 1) {
      float ov = __shfl_xor(bv, off, 32);
      int   oi = __shfl_xor(bi, off, 32);
      if (ov > bv || (ov == bv && oi < bi)) { bv = ov; bi = oi; }
    }
    __syncthreads();
    if ((tid & 31) == 0) { s_val[tid >> 5] = bv; s_idx[tid >> 5] = bi; }
    __syncthreads();
    if (tid < 32) {
      bv = s_val[tid]; bi = s_idx[tid];
#pragma unroll
      for (int off = 16; off; off >>= 1) {
        float ov = __shfl_xor(bv, off, 32);
        int   oi = __shfl_xor(bi, off, 32);
        if (ov > bv || (ov == bv && oi < bi)) { bv = ov; bi = oi; }
      }
      if (tid == 0) { s_last = bi; fpsInds[b * NPOINT + step] = bi; }
    }
    __syncthreads();
    int last = s_last;
#pragma unroll
    for (int j = 0; j < 16; ++j)
      if (tid * 16 + j == last) { s_c[0] = px[j]; s_c[1] = py[j]; s_c[2] = pz[j]; }
    __syncthreads();
  }
}

// ---------------------------------------------------------------------------
// Ball query: one wave32 per center; ballot/popc compaction keeps index order
// (== reference's sorted-candidate-take-16 semantics). Pads with first hit.
// ---------------------------------------------------------------------------
__global__ __launch_bounds__(256) void ballquery_kernel(const float* __restrict__ xyz,
                                                        const int* __restrict__ fpsInds,
                                                        int* __restrict__ ballIdx,
                                                        float* __restrict__ centers) {
  const int gid  = blockIdx.x * 8 + (threadIdx.x >> 5);   // center id 0..8191
  const int lane = threadIdx.x & 31;
  const int b    = gid >> 10;
  const float* P = xyz + (size_t)b * NPTS * 3;

  int ci = fpsInds[gid];
  float cx = P[ci * 3 + 0], cy = P[ci * 3 + 1], cz = P[ci * 3 + 2];
  if (lane == 0) {
    centers[gid * 3 + 0] = cx;
    centers[gid * 3 + 1] = cy;
    centers[gid * 3 + 2] = cz;
  }

  int cnt = 0, firstIdx = -1;
  for (int base = 0; base < NPTS && cnt < NSAMP; base += 32) {
    int pi = base + lane;
    float dx = P[pi * 3 + 0] - cx;
    float dy = P[pi * 3 + 1] - cy;
    float dz = P[pi * 3 + 2] - cz;
    bool in = (dx * dx + dy * dy + dz * dz) < RAD2;
    unsigned m = (unsigned)__ballot(in);
    if (m) {
      if (firstIdx < 0) firstIdx = base + (__ffs((int)m) - 1);
      int rank = __popc(m & ((1u << lane) - 1u));
      if (in && (cnt + rank) < NSAMP) ballIdx[gid * NSAMP + cnt + rank] = pi;
      cnt += __popc(m);
    }
  }
  if (cnt > NSAMP) cnt = NSAMP;
  if (lane < NSAMP && lane >= cnt) ballIdx[gid * NSAMP + lane] = firstIdx;
}

// ---------------------------------------------------------------------------
// Build X0 [131072][288] bf16 N-major: 3 normalized xyz + 256 gathered feats
// + zero pad. One thread per 8-channel chunk -> one b128 store.
// ---------------------------------------------------------------------------
__global__ __launch_bounds__(256) void buildx0_kernel(const float* __restrict__ xyz,
                                                      const float* __restrict__ features,
                                                      const int* __restrict__ ballIdx,
                                                      const float* __restrict__ centers,
                                                      unsigned short* __restrict__ X0) {
  int t = blockIdx.x * 256 + threadIdx.x;
  if (t >= N0 * (K0PAD / 8)) return;
  int n = t / (K0PAD / 8);
  int g = t % (K0PAD / 8);
  int b  = n >> 14;           // / (1024*16)
  int bp = n >> 4;            // b*1024 + p
  int idx = ballIdx[n];
  const float* C = centers + (size_t)bp * 3;
  const float* P = xyz + ((size_t)b * NPTS + idx) * 3;
  const float* F = features + (size_t)b * 256 * NPTS + idx;

  union { unsigned short us[8]; u32x4 v; } o;
#pragma unroll
  for (int j = 0; j < 8; ++j) {
    int c = g * 8 + j;
    float v;
    if (c < 3)        v = (P[c] - C[c]) * (1.0f / 0.3f);
    else if (c < 259) v = F[(size_t)(c - 3) * NPTS];
    else              v = 0.f;
    o.us[j] = f2bf(v);
  }
  *(u32x4*)(X0 + (size_t)n * K0PAD + g * 8) = o.v;
}

// ---------------------------------------------------------------------------
// Weight prep: f32 [M][Kin] -> bf16 [M][K0] (zero padded K).
// ---------------------------------------------------------------------------
__global__ __launch_bounds__(256) void prep_w_kernel(const float* __restrict__ w,
                                                     unsigned short* __restrict__ out,
                                                     int M, int Kin, int Kp) {
  int t = blockIdx.x * 256 + threadIdx.x;
  if (t >= M * Kp) return;
  int m = t / Kp, k = t % Kp;
  out[t] = (k < Kin) ? f2bf(w[(size_t)m * Kin + k]) : (unsigned short)0;
}

// ---------------------------------------------------------------------------
// bf16 WMMA GEMM:  Y[n][m] = sum_k W[m][k] * X[n][k]  (+bias)
//   W: [MDIM][KK] bf16 row-major.  X: [N][KK] bf16 N-major.
//   MODE 0: Y f32 [N][MDIM]         (pre-BN buffer)
//   MODE 1: Y f32 [B][MDIM][1024]   (direct transposed final output)
// 256 threads = 8 waves; workgroup tile MDIM x COLS. K-loop fully unrolled.
// Double-buffered LDS B-tile, ONE barrier per K-step; next tile's global
// loads are issued after the barrier so they overlap this step's WMMAs and
// are only waited at the next barrier.
// Per K-step per wave: 2 global b128 (A), 8 LDS b128 (B), 4 WMMAs.
// ---------------------------------------------------------------------------
template <int KK, int COLS, int MODE, int MDIM>
__global__ __launch_bounds__(256) void gemm_bf16_kernel(const unsigned short* __restrict__ W,
                                                        const unsigned short* __restrict__ X,
                                                        float* __restrict__ Y,
                                                        const float* __restrict__ bias) {
  constexpr int NSTEP = KK / 32;
  constexpr int CH    = (COLS * 4) / 256;   // 16B chunks per thread (1 or 2)
  constexpr int WPM   = MDIM / 16;          // waves per M strip (8 or 4)
  constexpr int BUFSZ = COLS * 40;          // 40-hword row stride: conflict-free

  __shared__ unsigned short Bt[2][BUFSZ];

  const int tid  = threadIdx.x;
  const int wid  = tid >> 5;
  const int lane = tid & 31;
  const int hf   = (lane >> 4) & 1;         // K-half select per WMMA layout
  const int l16  = lane & 15;
  const int mstrip = wid % WPM;
  const int cgrp   = wid / WPM;             // 0 (COLS=64) or 0/1 (COLS=128)

  const int n0    = blockIdx.x * COLS;
  const int mrow  = (mstrip << 4) + l16;         // A row for this lane
  const int mbase = (mstrip << 4) + (hf << 3);   // first of 8 D rows (lane)

  // Per-thread B-tile load slots, computed once (inner loop is pure adds).
  const unsigned short* gsrc[CH];
  unsigned short*       ldst[CH];
#pragma unroll
  for (int i = 0; i < CH; ++i) {
    int ch = tid + i * 256;
    int col = ch >> 2, part = ch & 3;
    gsrc[i] = X + (size_t)(n0 + col) * KK + part * 8;
    ldst[i] = &Bt[0][0] + col * 40 + part * 8;
  }
  const unsigned short* aptr = W + (size_t)mrow * KK + (hf << 3);

  v8f acc[4] = {};

  // prologue: tile 0 into buffer 0
#pragma unroll
  for (int i = 0; i < CH; ++i)
    *(u32x4*)(ldst[i]) = *(const u32x4*)(gsrc[i]);

#pragma unroll
  for (int s = 0; s < NSTEP; ++s) {
    __syncthreads();   // tile s visible; compute(s-1) done -> buffer (s+1)&1 free
    if (s + 1 < NSTEP) {
#pragma unroll
      for (int i = 0; i < CH; ++i)
        *(u32x4*)(ldst[i] + ((s + 1) & 1) * BUFSZ) =
            *(const u32x4*)(gsrc[i] + (s + 1) * 32);
    }
    // A fragment: 16x32 bf16; lane<16 holds K[0..7],[16..23]; lane>=16: +8.
    u32x4 a0 = *(const u32x4*)(aptr + s * 32);
    u32x4 a1 = *(const u32x4*)(aptr + s * 32 + 16);
    v16bf a = make_v16bf(a0, a1);
    const unsigned short* bbase = &Bt[s & 1][0];
#pragma unroll
    for (int j = 0; j < 4; ++j) {
      int col = (cgrp << 6) + (j << 4) + l16;
      u32x4 b0 = *(const u32x4*)(bbase + col * 40 + (hf << 3));
      u32x4 b1 = *(const u32x4*)(bbase + col * 40 + 16 + (hf << 3));
      v16bf bfr = make_v16bf(b0, b1);
      acc[j] = __builtin_amdgcn_wmma_f32_16x16x32_bf16(
          false, a, false, bfr, (short)0, acc[j], false, false);
    }
  }

  if (bias) {
#pragma unroll
    for (int r = 0; r < 8; ++r) {
      float bv = bias[mbase + r];
#pragma unroll
      for (int j = 0; j < 4; ++j) acc[j][r] += bv;
    }
  }

  if constexpr (MODE == 0) {
#pragma unroll
    for (int j = 0; j < 4; ++j) {
      int n = n0 + (cgrp << 6) + (j << 4) + l16;
      float* dst = Y + (size_t)n * MDIM + mbase;
      union { v8f v; f32x4 q[2]; } t;
      t.v = acc[j];
      *(f32x4*)dst       = t.q[0];
      *(f32x4*)(dst + 4) = t.q[1];
    }
  } else {
#pragma unroll
    for (int j = 0; j < 4; ++j) {
      int n  = n0 + (cgrp << 6) + (j << 4) + l16;
      int bb = n >> 10, p = n & 1023;
#pragma unroll
      for (int r = 0; r < 8; ++r)
        Y[((size_t)bb * MDIM + mbase + r) * 1024 + p] = acc[j][r];
    }
  }
}

// ---------------------------------------------------------------------------
// Deterministic BN stats: pass 1 per-1024-row partial sums/sumsq per channel.
// ---------------------------------------------------------------------------
__global__ __launch_bounds__(256) void stats_part_kernel(const float* __restrict__ Y,
                                                         float* __restrict__ part) {
  __shared__ float sS[256], sQ[256];
  int c = threadIdx.x & 127;
  int half = threadIdx.x >> 7;
  size_t base = ((size_t)blockIdx.x * 1024 + half * 512) * 128 + c;
  float s = 0.f, q = 0.f;
  for (int i = 0; i < 512; ++i) {
    float v = Y[base + (size_t)i * 128];
    s += v; q += v * v;
  }
  sS[threadIdx.x] = s; sQ[threadIdx.x] = q;
  __syncthreads();
  if (threadIdx.x < 128) {
    part[blockIdx.x * 256 + threadIdx.x]       = sS[threadIdx.x] + sS[threadIdx.x + 128];
    part[blockIdx.x * 256 + 128 + threadIdx.x] = sQ[threadIdx.x] + sQ[threadIdx.x + 128];
  }
}

// Pass 2: combine partials in fixed order, emit BN scale/shift.
__global__ void bn_finalize_kernel(const float* __restrict__ part, int G,
                                   const float* __restrict__ gam,
                                   const float* __restrict__ bet,
                                   float* __restrict__ scsh, float invCnt) {
  int t = threadIdx.x;           // 128
  float s = 0.f, q = 0.f;
  for (int wg = 0; wg < G; ++wg) {
    s += part[wg * 256 + t];
    q += part[wg * 256 + 128 + t];
  }
  float mean = s * invCnt;
  float var  = q * invCnt - mean * mean;
  float inv  = rsqrtf(var + 1e-5f);
  float sc   = gam[t] * inv;
  scsh[t]       = sc;
  scsh[128 + t] = bet[t] - mean * sc;
}

// BN apply + ReLU + bf16 quantize (feeds next GEMM's B operand).
__global__ __launch_bounds__(256) void bn_relu_kernel(const float* __restrict__ Y,
                                                      const float* __restrict__ scsh,
                                                      unsigned short* __restrict__ act,
                                                      int total) {
  int t = blockIdx.x * 256 + threadIdx.x;
  if (t >= total) return;
  int m = t & 127;
  float v = Y[t] * scsh[m] + scsh[128 + m];
  act[t] = f2bf(v > 0.f ? v : 0.f);
}

// Max over NSAMP=16 samples: act [131072][128] bf16 -> feat [8192][128] bf16.
__global__ __launch_bounds__(256) void maxpool_kernel(const unsigned short* __restrict__ act,
                                                      unsigned short* __restrict__ feat) {
  int t = blockIdx.x * 256 + threadIdx.x;   // 8192*128
  int bp = t >> 7, c = t & 127;
  float m = -1e30f;
#pragma unroll
  for (int s = 0; s < NSAMP; ++s)
    m = fmaxf(m, bf2f(act[((size_t)(bp << 4) + s) * 128 + c]));
  feat[t] = f2bf(m);
}

// ---------------------------------------------------------------------------
extern "C" void kernel_launch(void* const* d_in, const int* in_sizes, int n_in,
                              void* d_out, int out_size, void* d_ws, size_t ws_size,
                              hipStream_t stream) {
  const float* xyz      = (const float*)d_in[0];
  const float* features = (const float*)d_in[1];
  const float* mlp_w0 = (const float*)d_in[2];
  const float* mlp_g0 = (const float*)d_in[3];
  const float* mlp_b0 = (const float*)d_in[4];
  const float* mlp_w1 = (const float*)d_in[5];
  const float* mlp_g1 = (const float*)d_in[6];
  const float* mlp_b1 = (const float*)d_in[7];
  const float* mlp_w2 = (const float*)d_in[8];
  const float* mlp_g2 = (const float*)d_in[9];
  const float* mlp_b2 = (const float*)d_in[10];
  const float* conv1_w = (const float*)d_in[11];
  const float* conv1_b = (const float*)d_in[12];
  const float* bn1_g   = (const float*)d_in[13];
  const float* bn1_b   = (const float*)d_in[14];
  const float* conv2_w = (const float*)d_in[15];
  const float* conv2_b = (const float*)d_in[16];
  const float* bn2_g   = (const float*)d_in[17];
  const float* bn2_b   = (const float*)d_in[18];
  const float* conv3_w = (const float*)d_in[19];
  const float* conv3_b = (const float*)d_in[20];
  float* out = (float*)d_out;

  // ---- workspace carve-up (256B aligned) ----
  char* w = (char*)d_ws;
  auto carve = [&](size_t bytes) {
    char* p = w;
    w += (bytes + 255) & ~size_t(255);
    return p;
  };
  unsigned short* X0    = (unsigned short*)carve((size_t)N0 * K0PAD * 2);  // 75.5 MB
  float*          yws   = (float*)carve((size_t)N0 * 128 * 4);             // 67.1 MB
  unsigned short* actA  = (unsigned short*)carve((size_t)N0 * 128 * 2);    // 33.6 MB
  unsigned short* actB  = (unsigned short*)carve((size_t)N0 * 128 * 2);    // 33.6 MB
  unsigned short* feat0 = (unsigned short*)carve((size_t)NPQ * 128 * 2);
  unsigned short* feat1 = (unsigned short*)carve((size_t)NPQ * 128 * 2);
  unsigned short* feat2 = (unsigned short*)carve((size_t)NPQ * 128 * 2);
  unsigned short* w0p = (unsigned short*)carve(128 * K0PAD * 2);
  unsigned short* w1p = (unsigned short*)carve(128 * 128 * 2);
  unsigned short* w2p = (unsigned short*)carve(128 * 128 * 2);
  unsigned short* c1p = (unsigned short*)carve(128 * 128 * 2);
  unsigned short* c2p = (unsigned short*)carve(128 * 128 * 2);
  unsigned short* c3p = (unsigned short*)carve(64 * 128 * 2);
  float* part    = (float*)carve(128 * 256 * 4);   // stats partials (max G=128)
  float* scsh    = (float*)carve(256 * 4);
  int*   fpsInds = (int*)carve((size_t)NPQ * 4);
  float* centers = (float*)carve((size_t)NPQ * 3 * 4);
  int*   ballIdx = (int*)carve((size_t)NPQ * NSAMP * 4);

  // ---- weight prep ----
  prep_w_kernel<<<(128 * K0PAD + 255) / 256, 256, 0, stream>>>(mlp_w0, w0p, 128, 259, K0PAD);
  prep_w_kernel<<<(128 * 128 + 255) / 256, 256, 0, stream>>>(mlp_w1, w1p, 128, 128, 128);
  prep_w_kernel<<<(128 * 128 + 255) / 256, 256, 0, stream>>>(mlp_w2, w2p, 128, 128, 128);
  prep_w_kernel<<<(128 * 128 + 255) / 256, 256, 0, stream>>>(conv1_w, c1p, 128, 128, 128);
  prep_w_kernel<<<(128 * 128 + 255) / 256, 256, 0, stream>>>(conv2_w, c2p, 128, 128, 128);
  prep_w_kernel<<<(64 * 128 + 255) / 256, 256, 0, stream>>>(conv3_w, c3p, 64, 128, 128);

  // ---- sampling / grouping ----
  fps_kernel<<<NBATCH, 1024, 0, stream>>>(xyz, fpsInds);
  ballquery_kernel<<<NPQ / 8, 256, 0, stream>>>(xyz, fpsInds, ballIdx, centers);
  buildx0_kernel<<<(N0 * (K0PAD / 8)) / 256, 256, 0, stream>>>(xyz, features, ballIdx, centers, X0);

  // ---- MLP layer 0: X0(288) -> actA ----
  gemm_bf16_kernel<K0PAD, 64, 0, 128><<<N0 / 64, 256, 0, stream>>>(w0p, X0, yws, nullptr);
  stats_part_kernel<<<N0 / 1024, 256, 0, stream>>>(yws, part);
  bn_finalize_kernel<<<1, 128, 0, stream>>>(part, N0 / 1024, mlp_g0, mlp_b0, scsh, 1.0f / N0);
  bn_relu_kernel<<<(N0 * 128) / 256, 256, 0, stream>>>(yws, scsh, actA, N0 * 128);

  // ---- MLP layer 1: actA -> actB ----
  gemm_bf16_kernel<128, 64, 0, 128><<<N0 / 64, 256, 0, stream>>>(w1p, actA, yws, nullptr);
  stats_part_kernel<<<N0 / 1024, 256, 0, stream>>>(yws, part);
  bn_finalize_kernel<<<1, 128, 0, stream>>>(part, N0 / 1024, mlp_g1, mlp_b1, scsh, 1.0f / N0);
  bn_relu_kernel<<<(N0 * 128) / 256, 256, 0, stream>>>(yws, scsh, actB, N0 * 128);

  // ---- MLP layer 2: actB -> actA ----
  gemm_bf16_kernel<128, 64, 0, 128><<<N0 / 64, 256, 0, stream>>>(w2p, actB, yws, nullptr);
  stats_part_kernel<<<N0 / 1024, 256, 0, stream>>>(yws, part);
  bn_finalize_kernel<<<1, 128, 0, stream>>>(part, N0 / 1024, mlp_g2, mlp_b2, scsh, 1.0f / N0);
  bn_relu_kernel<<<(N0 * 128) / 256, 256, 0, stream>>>(yws, scsh, actA, N0 * 128);

  // ---- max over samples ----
  maxpool_kernel<<<(NPQ * 128) / 256, 256, 0, stream>>>(actA, feat0);

  // ---- conv1 (+bias) -> BN -> ReLU ----
  gemm_bf16_kernel<128, 64, 0, 128><<<NPQ / 64, 256, 0, stream>>>(c1p, feat0, yws, conv1_b);
  stats_part_kernel<<<NPQ / 1024, 256, 0, stream>>>(yws, part);
  bn_finalize_kernel<<<1, 128, 0, stream>>>(part, NPQ / 1024, bn1_g, bn1_b, scsh, 1.0f / NPQ);
  bn_relu_kernel<<<(NPQ * 128) / 256, 256, 0, stream>>>(yws, scsh, feat1, NPQ * 128);

  // ---- conv2 (+bias) -> BN -> ReLU ----
  gemm_bf16_kernel<128, 64, 0, 128><<<NPQ / 64, 256, 0, stream>>>(c2p, feat1, yws, conv2_b);
  stats_part_kernel<<<NPQ / 1024, 256, 0, stream>>>(yws, part);
  bn_finalize_kernel<<<1, 128, 0, stream>>>(part, NPQ / 1024, bn2_g, bn2_b, scsh, 1.0f / NPQ);
  bn_relu_kernel<<<(NPQ * 128) / 256, 256, 0, stream>>>(yws, scsh, feat2, NPQ * 128);

  // ---- conv3 (+bias) -> d_out (8,64,1024) transposed write ----
  gemm_bf16_kernel<128, 128, 1, 64><<<NPQ / 128, 256, 0, stream>>>(c3p, feat2, out, conv3_b);
}
